// CausalGQAttention_1803886264590
// MI455X (gfx1250) — matrix operand order repaired and checked by reading, not compiled, not run on hardware
//
#include <hip/hip_runtime.h>
#include <hip/hip_bf16.h>

// ---------------------------------------------------------------------------
// CDNA5 (gfx1250) causal GQA prefill, bf16 WMMA pipeline.
// Reference: out = (attn(rope(xWq), rope(xWk), xWv)) @ Wo ; returns (out, k, v)
// B=2 T=2048 D=2048 NH=32 NKV=8 HD=64
//
// Compute-bound (~155 GFLOP vs ~5us of HBM traffic at 23.3 TB/s), so all
// matmuls run through v_wmma_f32_16x16x32_bf16. GEMM uses a double-buffered
// K=64 LDS stage (one barrier per 32 wmma per wave) to overlap global loads
// with matrix ops.
// ---------------------------------------------------------------------------

typedef __attribute__((ext_vector_type(16))) __bf16 v16bf;
typedef __attribute__((ext_vector_type(8)))  float  v8f;

#define Bn   2
#define Tn   2048
#define Dm   2048
#define NH   32
#define NKV  8
#define HD   64
#define BT   (Bn * Tn)            // 4096

union Frag {                       // one WMMA bf16 A/B operand (16 bf16 / lane)
    v16bf v;
    uint4 q[2];
    __bf16 h[16];
};
union U8 {                         // 8 bf16 <-> uint4 repack helper
    uint4 q;
    __bf16 h[8];
};

__device__ inline __bf16 f2bf(float f) {
    unsigned int u = __builtin_bit_cast(unsigned int, f);
    unsigned int r = (u + 0x7FFFu + ((u >> 16) & 1u)) >> 16;
    return __builtin_bit_cast(__bf16, (unsigned short)r);
}

__device__ inline v8f wmma_bf16(const v16bf a, const v16bf b, v8f c) {
    // D = A(16x32 bf16) * B(32x16 bf16) + C(16x16 f32)
    return __builtin_amdgcn_wmma_f32_16x16x32_bf16(
        /*neg_a=*/false, a, /*neg_b=*/false, b,
        /*c_mod=*/(short)0, c, /*reuse_a=*/false, /*reuse_b=*/false);
}

// ---------------------------------------------------------------------------
// fp32 -> bf16 pack
// ---------------------------------------------------------------------------
__global__ __launch_bounds__(256)
void pack_bf16(const float* __restrict__ src, __bf16* __restrict__ dst, int n) {
    int i = blockIdx.x * 256 + threadIdx.x;
    if (i < n) dst[i] = f2bf(src[i]);
}

// ---------------------------------------------------------------------------
// Generic C(f32, MxN) = A(bf16, MxK row-major) * B(bf16, KxN row-major)
// 128x128 block tile, 4 waves of 64x64.
// Software pipeline: K staged in chunks of 64, double-buffered LDS; next
// tile's global loads issued before the wmma block, stored to the ping-pong
// buffer after, single barrier per stage.
// A-fragment lane layout (ISA 7.12.2, 16-bit A 16x32):
//   lane l: row = l&15, khalf = l>>4 ; elem j: K = (j/8)*16 + khalf*8 + (j%8)
// B-fragment (32x16): lane l: col = l&15, khalf = l>>4 ; elem j: K = (j/8)*16 + khalf*8 + (j%8)
// B is staged transposed in LDS so fragments are contiguous b128 loads.
// ---------------------------------------------------------------------------
__global__ __launch_bounds__(128)
void gemm_bf16_wmma(const __bf16* __restrict__ A, const __bf16* __restrict__ B,
                    float* __restrict__ C, int M, int N, int K) {
    __shared__ __bf16 As[2][128][72];   // [buf][m][k], 144B rows (16B aligned)
    __shared__ __bf16 Bst[2][128][72];  // [buf][n][k] (transposed)

    const int tid  = threadIdx.x;
    const int lane = tid & 31;
    const int w    = tid >> 5;
    const int kh   = lane >> 4;
    const int l15  = lane & 15;
    const int rowBase = blockIdx.y * 128;
    const int colBase = blockIdx.x * 128;
    const int waveRow = (w >> 1) * 64;
    const int waveCol = (w & 1) * 64;

    const int aRow = rowBase + tid;     // 1 A-row per thread, 64 bf16 / stage
    const int bKr  = tid >> 1;          // B k-row, 64 bf16 per (thread, half)
    const int bNb  = (tid & 1) * 64;

    v8f acc[4][4];
#pragma unroll
    for (int mt = 0; mt < 4; ++mt)
#pragma unroll
        for (int nt = 0; nt < 4; ++nt) acc[mt][nt] = {};

    uint4 ra[8], rb[8];
    auto gload = [&](int kb) {
        const uint4* pa = (const uint4*)(A + (size_t)aRow * K + kb);
#pragma unroll
        for (int c = 0; c < 8; ++c) ra[c] = pa[c];
        const uint4* pb = (const uint4*)(B + (size_t)(kb + bKr) * N + colBase + bNb);
#pragma unroll
        for (int c = 0; c < 8; ++c) rb[c] = pb[c];
    };
    auto lstore = [&](int buf) {
        uint4* da = (uint4*)&As[buf][tid][0];
#pragma unroll
        for (int c = 0; c < 8; ++c) da[c] = ra[c];
#pragma unroll
        for (int c = 0; c < 8; ++c) {
            U8 f; f.q = rb[c];
#pragma unroll
            for (int e = 0; e < 8; ++e) Bst[buf][bNb + c * 8 + e][bKr] = f.h[e];
        }
    };

    gload(0);
    lstore(0);
    __syncthreads();

    for (int kb = 0; kb < K; kb += 64) {
        const int  buf  = (kb >> 6) & 1;
        const bool more = (kb + 64) < K;
        if (more) gload(kb + 64);       // overlap with wmma below

#pragma unroll
        for (int kc = 0; kc < 2; ++kc) {
            Frag afr[4], bfr[4];
#pragma unroll
            for (int mt = 0; mt < 4; ++mt) {
                const int r = waveRow + mt * 16 + l15;
                afr[mt].q[0] = *(const uint4*)&As[buf][r][kc * 32 + kh * 8];
                afr[mt].q[1] = *(const uint4*)&As[buf][r][kc * 32 + 16 + kh * 8];
            }
#pragma unroll
            for (int nt = 0; nt < 4; ++nt) {
                const int n = waveCol + nt * 16 + l15;
                bfr[nt].q[0] = *(const uint4*)&Bst[buf][n][kc * 32 + kh * 8];
                bfr[nt].q[1] = *(const uint4*)&Bst[buf][n][kc * 32 + 16 + kh * 8];
            }
#pragma unroll
            for (int mt = 0; mt < 4; ++mt)
#pragma unroll
                for (int nt = 0; nt < 4; ++nt)
                    acc[mt][nt] = wmma_bf16(afr[mt].v, bfr[nt].v, acc[mt][nt]);
        }

        if (more) lstore(buf ^ 1);      // buf^1 last read in previous stage, before its barrier; safe
        __syncthreads();
    }

    // C layout: lane l, vgpr i -> m = (l>>4)*8 + i, n = l&15.
    // Base pointer per lane; nt offsets are small immediates.
    float* Cp = C + (size_t)(rowBase + waveRow + kh * 8) * N
                  + (colBase + waveCol + l15);
#pragma unroll
    for (int mt = 0; mt < 4; ++mt)
#pragma unroll
        for (int i = 0; i < 8; ++i) {
            float* rowp = Cp + (size_t)(mt * 16 + i) * N;
#pragma unroll
            for (int nt = 0; nt < 4; ++nt)
                rowp[nt * 16] = acc[mt][nt][i];
        }
}

// ---------------------------------------------------------------------------
// RoPE on Q,K (fp32), emit fp32 k/v tuple outputs + bf16 copies for attention.
// Kr/Vb layout: [b][g][s][d]  (g = kv head)
// ---------------------------------------------------------------------------
__global__ __launch_bounds__(256)
void rope_split(const float* __restrict__ Qf, const float* __restrict__ Kf,
                const float* __restrict__ Vf,
                __bf16* __restrict__ Qr, __bf16* __restrict__ Krb,
                __bf16* __restrict__ Vbb,
                float* __restrict__ Kout, float* __restrict__ Vout) {
    const int bt = blockIdx.x;                 // 0..BT-1
    const int b = bt >> 11, t = bt & (Tn - 1);
    const int tid = threadIdx.x;
    const float ln_base = __logf(10000.0f);

    // Q: 32 heads x 32 rotation pairs
    for (int idx = tid; idx < NH * 32; idx += 256) {
        const int h = idx >> 5, d = idx & 31;
        const float ang = (float)t * __expf(-(float)d * (1.0f / 32.0f) * ln_base);
        const float c = __cosf(ang), s = __sinf(ang);
        const size_t base = (size_t)bt * Dm + h * HD + d;
        const float q0 = Qf[base], q1 = Qf[base + 32];
        Qr[base]      = f2bf(q0 * c - q1 * s);
        Qr[base + 32] = f2bf(q1 * c + q0 * s);
    }
    // K: 8 kv heads x 32 pairs
    for (int idx = tid; idx < NKV * 32; idx += 256) {
        const int g = idx >> 5, d = idx & 31;
        const float ang = (float)t * __expf(-(float)d * (1.0f / 32.0f) * ln_base);
        const float c = __cosf(ang), s = __sinf(ang);
        const size_t src = (size_t)bt * (NKV * HD) + g * HD + d;
        const float k0 = Kf[src], k1 = Kf[src + 32];
        const float r0 = k0 * c - k1 * s;
        const float r1 = k1 * c + k0 * s;
        const size_t dst = (((size_t)(b * NKV + g)) * Tn + t) * HD + d;
        Kout[dst]      = r0;      Kout[dst + 32] = r1;
        Krb[dst]       = f2bf(r0); Krb[dst + 32] = f2bf(r1);
    }
    // V: straight copy / pack
    for (int idx = tid; idx < NKV * HD; idx += 256) {
        const int g = idx >> 6, d = idx & 63;
        const size_t src = (size_t)bt * (NKV * HD) + g * HD + d;
        const float v = Vf[src];
        const size_t dst = (((size_t)(b * NKV + g)) * Tn + t) * HD + d;
        Vout[dst] = v;
        Vbb[dst]  = f2bf(v);
    }
}

// ---------------------------------------------------------------------------
// Flash-style causal GQA attention.
// Block = (b, h, 64-query-row block); 4 waves, each owns 16 query rows.
// S processed in shared 32-row chunks (K/V staged in LDS by all 128 threads).
// Per chunk per wave: 2x2 QK^T wmma (K=32 each half of hd) + 4 PV wmma.
// ---------------------------------------------------------------------------
__global__ __launch_bounds__(128)
void attn_gqa(const __bf16* __restrict__ Qr, const __bf16* __restrict__ Kr,
              const __bf16* __restrict__ Vb, __bf16* __restrict__ AO) {
    __shared__ __bf16 Kc[32][64];      // [s_local][d]
    __shared__ __bf16 Vct[64][40];     // transposed: [d][s_local], padded
    __shared__ __bf16 Pl[4][16][32];   // per-wave P staging (C-layout -> A-layout)

    const int bx = blockIdx.x;
    const int qb = bx & 31;            // T/64 blocks
    const int h  = (bx >> 5) & 31;
    const int b  = bx >> 10;
    const int g  = h >> 2;             // kv group (N_REP = 4)
    const int tid = threadIdx.x, lane = tid & 31, w = tid >> 5;
    const int kh = lane >> 4, l15 = lane & 15;
    const int q0 = qb * 64 + w * 16;

    // ---- Q fragments (16 rows x hd=64 -> two K=32 A-frags) ----
    Frag aq0, aq1;
    {
        const __bf16* qrow = Qr + ((size_t)(b * Tn + q0 + l15)) * Dm + h * HD;
        aq0.q[0] = *(const uint4*)(qrow + kh * 8);
        aq0.q[1] = *(const uint4*)(qrow + 16 + kh * 8);
        aq1.q[0] = *(const uint4*)(qrow + 32 + kh * 8);
        aq1.q[1] = *(const uint4*)(qrow + 48 + kh * 8);
    }

    float mrow[8], lsum[8];
    v8f o[4];
#pragma unroll
    for (int i = 0; i < 8; ++i) { mrow[i] = -3.0e38f; lsum[i] = 0.0f; }
#pragma unroll
    for (int nt = 0; nt < 4; ++nt) o[nt] = {};

    const __bf16* Kbase = Kr + ((size_t)(b * NKV + g)) * Tn * HD;
    const __bf16* Vbase = Vb + ((size_t)(b * NKV + g)) * Tn * HD;
    const int smax = qb * 64 + 64;     // block-uniform causal bound (exclusive)

    for (int sb = 0; sb < smax; sb += 32) {
        // ---- cooperative stage: K chunk row-major, V chunk transposed ----
#pragma unroll
        for (int c = 0; c < 2; ++c) {
            const int idx = tid * 2 + c;         // 0..255
            const int r = idx >> 3, c8 = (idx & 7) * 8;
            *(uint4*)&Kc[r][c8] = *(const uint4*)(Kbase + (size_t)(sb + r) * HD + c8);
            U8 f;
            f.q = *(const uint4*)(Vbase + (size_t)(sb + r) * HD + c8);
#pragma unroll
            for (int e = 0; e < 8; ++e) Vct[c8 + e][r] = f.h[e];
        }
        __syncthreads();

        if (sb <= q0 + 15) {
            // ---- scores: two 16x16 tiles over this 32-wide S chunk ----
            v8f s0 = {}, s1 = {};
            {
                Frag bk;
                bk.q[0] = *(const uint4*)&Kc[l15][kh * 8];
                bk.q[1] = *(const uint4*)&Kc[l15][16 + kh * 8];
                s0 = wmma_bf16(aq0.v, bk.v, s0);
                bk.q[0] = *(const uint4*)&Kc[l15][32 + kh * 8];
                bk.q[1] = *(const uint4*)&Kc[l15][48 + kh * 8];
                s0 = wmma_bf16(aq1.v, bk.v, s0);
                bk.q[0] = *(const uint4*)&Kc[16 + l15][kh * 8];
                bk.q[1] = *(const uint4*)&Kc[16 + l15][16 + kh * 8];
                s1 = wmma_bf16(aq0.v, bk.v, s1);
                bk.q[0] = *(const uint4*)&Kc[16 + l15][32 + kh * 8];
                bk.q[1] = *(const uint4*)&Kc[16 + l15][48 + kh * 8];
                s1 = wmma_bf16(aq1.v, bk.v, s1);
            }

            // ---- scale + causal mask + online softmax ----
            float p0[8], p1[8], al[8];
#pragma unroll
            for (int i = 0; i < 8; ++i) {
                const int q = q0 + kh * 8 + i;
                float a = s0[i] * 0.125f;
                float c = s1[i] * 0.125f;
                if (sb + l15 > q)       a = -3.0e38f;
                if (sb + 16 + l15 > q)  c = -3.0e38f;
                p0[i] = a; p1[i] = c;
            }
#pragma unroll
            for (int i = 0; i < 8; ++i) {
                float tm = fmaxf(p0[i], p1[i]);
#pragma unroll
                for (int off = 1; off < 16; off <<= 1)
                    tm = fmaxf(tm, __shfl_xor(tm, off, 32));
                const float nm = fmaxf(mrow[i], tm);
                al[i] = __expf(mrow[i] - nm);
                p0[i] = __expf(p0[i] - nm);
                p1[i] = __expf(p1[i] - nm);
                float rs = p0[i] + p1[i];
#pragma unroll
                for (int off = 1; off < 16; off <<= 1)
                    rs += __shfl_xor(rs, off, 32);
                lsum[i] = lsum[i] * al[i] + rs;
                mrow[i] = nm;
            }
#pragma unroll
            for (int nt = 0; nt < 4; ++nt)
#pragma unroll
                for (int i = 0; i < 8; ++i) o[nt][i] *= al[i];

            // ---- stage P (C-layout -> A-layout via LDS; same-wave, in-order) ----
#pragma unroll
            for (int i = 0; i < 8; ++i) {
                Pl[w][kh * 8 + i][l15]      = f2bf(p0[i]);
                Pl[w][kh * 8 + i][16 + l15] = f2bf(p1[i]);
            }
            Frag pa;
            pa.q[0] = *(const uint4*)&Pl[w][l15][kh * 8];
            pa.q[1] = *(const uint4*)&Pl[w][l15][16 + kh * 8];

            // ---- O += P @ V ----
#pragma unroll
            for (int nt = 0; nt < 4; ++nt) {
                Frag bv;
                bv.q[0] = *(const uint4*)&Vct[nt * 16 + l15][kh * 8];
                bv.q[1] = *(const uint4*)&Vct[nt * 16 + l15][16 + kh * 8];
                o[nt] = wmma_bf16(pa.v, bv.v, o[nt]);
            }
        }
        __syncthreads();
    }

    // ---- finalize: O /= lsum, write bf16 attn-out [bt][h*64+d] ----
    float inv[8];
#pragma unroll
    for (int i = 0; i < 8; ++i) inv[i] = 1.0f / lsum[i];
#pragma unroll
    for (int nt = 0; nt < 4; ++nt)
#pragma unroll
        for (int i = 0; i < 8; ++i) {
            const size_t row = (size_t)(b * Tn + q0 + kh * 8 + i);
            AO[row * Dm + h * HD + nt * 16 + l15] = f2bf(o[nt][i] * inv[i]);
        }
}

// ---------------------------------------------------------------------------
// Host launch
// ---------------------------------------------------------------------------
extern "C" void kernel_launch(void* const* d_in, const int* in_sizes, int n_in,
                              void* d_out, int out_size, void* d_ws, size_t ws_size,
                              hipStream_t stream) {
    const float* x  = (const float*)d_in[0];
    const float* Wq = (const float*)d_in[1];
    const float* Wk = (const float*)d_in[2];
    const float* Wv = (const float*)d_in[3];
    const float* Wo = (const float*)d_in[4];

    float* out  = (float*)d_out;                       // [B,T,D]
    float* kout = out + (size_t)BT * Dm;               // [B,NKV,T,HD]
    float* vout = kout + (size_t)Bn * NKV * Tn * HD;   // [B,NKV,T,HD]

    // workspace carve-out (~130 MB)
    char* ws = (char*)d_ws;
    size_t off = 0;
    auto carve = [&](size_t bytes) {
        void* p = ws + off;
        off += (bytes + 255) & ~(size_t)255;
        return p;
    };
    __bf16* xb  = (__bf16*)carve((size_t)BT * Dm * 2);
    __bf16* wqb = (__bf16*)carve((size_t)Dm * Dm * 2);
    __bf16* wkb = (__bf16*)carve((size_t)Dm * (NKV * HD) * 2);
    __bf16* wvb = (__bf16*)carve((size_t)Dm * (NKV * HD) * 2);
    __bf16* wob = (__bf16*)carve((size_t)Dm * Dm * 2);
    float*  Qf  = (float*)carve((size_t)BT * Dm * 4);
    float*  Kf  = (float*)carve((size_t)BT * (NKV * HD) * 4);
    float*  Vf  = (float*)carve((size_t)BT * (NKV * HD) * 4);
    __bf16* Qr  = (__bf16*)carve((size_t)BT * Dm * 2);
    __bf16* Krb = (__bf16*)carve((size_t)Bn * NKV * Tn * HD * 2);
    __bf16* Vbb = (__bf16*)carve((size_t)Bn * NKV * Tn * HD * 2);
    __bf16* AO  = (__bf16*)carve((size_t)BT * Dm * 2);
    (void)ws_size; (void)n_in; (void)in_sizes; (void)out_size;

    // 1) pack to bf16
    pack_bf16<<<(BT * Dm + 255) / 256, 256, 0, stream>>>(x, xb, BT * Dm);
    pack_bf16<<<(Dm * Dm + 255) / 256, 256, 0, stream>>>(Wq, wqb, Dm * Dm);
    pack_bf16<<<(Dm * NKV * HD + 255) / 256, 256, 0, stream>>>(Wk, wkb, Dm * NKV * HD);
    pack_bf16<<<(Dm * NKV * HD + 255) / 256, 256, 0, stream>>>(Wv, wvb, Dm * NKV * HD);
    pack_bf16<<<(Dm * Dm + 255) / 256, 256, 0, stream>>>(Wo, wob, Dm * Dm);

    // 2) QKV projections
    gemm_bf16_wmma<<<dim3(Dm / 128, BT / 128), 128, 0, stream>>>(xb, wqb, Qf, BT, Dm, Dm);
    gemm_bf16_wmma<<<dim3((NKV * HD) / 128, BT / 128), 128, 0, stream>>>(xb, wkb, Kf, BT, NKV * HD, Dm);
    gemm_bf16_wmma<<<dim3((NKV * HD) / 128, BT / 128), 128, 0, stream>>>(xb, wvb, Vf, BT, NKV * HD, Dm);

    // 3) RoPE + tuple outputs (k, v) + bf16 repack for attention
    rope_split<<<BT, 256, 0, stream>>>(Qf, Kf, Vf, Qr, Krb, Vbb, kout, vout);

    // 4) causal grouped attention
    attn_gqa<<<Bn * NH * (Tn / 64), 128, 0, stream>>>(Qr, Krb, Vbb, AO);

    // 5) output projection
    gemm_bf16_wmma<<<dim3(Dm / 128, BT / 128), 128, 0, stream>>>(AO, wob, out, BT, Dm, Dm);
}
